// LSTM_4518305595857
// MI455X (gfx1250) — matrix operand order repaired and checked by reading, not compiled, not run on hardware
//
#include <hip/hip_runtime.h>
#include <stdint.h>

#define T_STEPS 1024
#define BATCH   64
#define ISZ     512
#define HSZ     512
#define GSZ     2048   // 4*H

#ifndef __has_builtin
#define __has_builtin(x) 0
#endif

#if __has_builtin(__builtin_amdgcn_global_load_async_to_lds_b128)
#define HAVE_ASYNC_LDS 1
#else
#define HAVE_ASYNC_LDS 0
#endif

#if __has_builtin(__builtin_amdgcn_s_cluster_barrier)
#define HAVE_CLUSTER_BARRIER 1
#else
#define HAVE_CLUSTER_BARRIER 0
#endif

typedef __attribute__((ext_vector_type(16))) __bf16 v16bf;
typedef __attribute__((ext_vector_type(8)))  float  v8f;

union Frag16 {
    v16bf v;
    uint4 u[2];
    unsigned short s[16];
};

__device__ __forceinline__ unsigned short f32_to_bf16(float f) {
    union { float f; unsigned int u; } x; x.f = f;
    unsigned int r = x.u + 0x7fffu + ((x.u >> 16) & 1u);  // round-to-nearest-even
    return (unsigned short)(r >> 16);
}

__device__ __forceinline__ float fsigmoid(float x) {
    return 1.0f / (1.0f + __expf(-x));
}
__device__ __forceinline__ float ftanh(float x) {
    float e = __expf(-2.0f * x);
    return (1.0f - e) / (1.0f + e);
}

#if HAVE_ASYNC_LDS
// Parameter type of the async builtin on this toolchain: vector-of-4-int
// pointers, global (AS1) source / LDS (AS3) destination.
typedef int vec4i __attribute__((vector_size(16)));
typedef __attribute__((address_space(1))) vec4i* g_v4i_ptr;
typedef __attribute__((address_space(3))) vec4i* l_v4i_ptr;
#endif

// 16-byte global -> LDS copy. On gfx1250 use the async direct-to-LDS path
// (no VGPR round trip, tracked by ASYNCcnt); otherwise plain load/store.
__device__ __forceinline__ void copy16_g2l(const void* g, void* l) {
#if HAVE_ASYNC_LDS
    __builtin_amdgcn_global_load_async_to_lds_b128(
        (g_v4i_ptr)g, (l_v4i_ptr)l, 0, 0);
#else
    *(uint4*)l = *(const uint4*)g;
#endif
}

__device__ __forceinline__ void g2l_wait() {
#if HAVE_ASYNC_LDS
#if __has_builtin(__builtin_amdgcn_s_wait_asynccnt)
    __builtin_amdgcn_s_wait_asynccnt(0);
#else
    asm volatile("s_wait_asynccnt 0x0" ::: "memory");
#endif
#endif
}

// ---------------------------------------------------------------------------
// Kernel A: f32 -> bf16 conversion (grid-stride)
// ---------------------------------------------------------------------------
__global__ void convert_f32_bf16_kernel(const float* __restrict__ src,
                                        unsigned short* __restrict__ dst, int n) {
    int i = blockIdx.x * blockDim.x + threadIdx.x;
    int stride = gridDim.x * blockDim.x;
    for (; i < n; i += stride) dst[i] = f32_to_bf16(src[i]);
}

// ---------------------------------------------------------------------------
// Kernel B: x_gates = x @ Wih^T + bias_ih + bias_hh
// M = T*B = 65536, N = 4H = 2048, K = I = 512. Block tile 128x128, 8 waves,
// wave tile 64(M) x 32(N) = 4 m-subtiles x 2 n-subtiles of v_wmma 16x16x32.
// Fully parallel (8192 blocks) -> occupancy hides global latency; x_bf (64MB)
// and Wih (2MB) stay resident in the 192MB L2 across tile re-reads.
// ---------------------------------------------------------------------------
__global__ __launch_bounds__(256) void xgates_gemm_kernel(
    const unsigned short* __restrict__ xb,    // [T*B][ISZ] bf16
    const unsigned short* __restrict__ wb,    // [GSZ][ISZ] bf16 (row = gate col)
    const float* __restrict__ bias_ih,
    const float* __restrict__ bias_hh,
    float* __restrict__ xg)                   // [T*B][GSZ] f32
{
    const int lane = threadIdx.x & 31;
    const int wv   = threadIdx.x >> 5;
    const int wm   = blockIdx.x * 128 + (wv & 1) * 64;
    const int wn   = blockIdx.y * 128 + (wv >> 1) * 32;
    const int lh   = lane >> 4;   // k-half select
    const int ll   = lane & 15;   // row/col within tile

    v8f acc[4][2] = {};

    for (int ks = 0; ks < 16; ++ks) {
        const int kb = ks * 32;
        Frag16 bfr[2];
#pragma unroll
        for (int nt = 0; nt < 2; ++nt) {
            // B fragment: lane holds W row (wn+nt*16+ll), K = kb + lh*16 + [0..15]
            const size_t base = (size_t)(wn + nt * 16 + ll) * ISZ + kb + lh * 16;
            bfr[nt].u[0] = *(const uint4*)(wb + base);
            bfr[nt].u[1] = *(const uint4*)(wb + base + 8);
        }
#pragma unroll
        for (int mt = 0; mt < 4; ++mt) {
            // A fragment: lane holds x row (wm+mt*16+ll), K = kb + lh*16 + [0..15]
            Frag16 a;
            const size_t b0 = (size_t)(wm + mt * 16 + ll) * ISZ + kb + lh * 16;
            a.u[0] = *(const uint4*)(xb + b0);
            a.u[1] = *(const uint4*)(xb + b0 + 8);
            acc[mt][0] = __builtin_amdgcn_wmma_f32_16x16x32_bf16(
                false, a.v, false, bfr[0].v, (short)0, acc[mt][0], false, false);
            acc[mt][1] = __builtin_amdgcn_wmma_f32_16x16x32_bf16(
                false, a.v, false, bfr[1].v, (short)0, acc[mt][1], false, false);
        }
    }

#pragma unroll
    for (int nt = 0; nt < 2; ++nt) {
        const int col = wn + nt * 16 + ll;
        const float bsum = bias_ih[col] + bias_hh[col];
#pragma unroll
        for (int mt = 0; mt < 4; ++mt) {
#pragma unroll
            for (int r = 0; r < 8; ++r) {
                const size_t row = (size_t)(wm + mt * 16 + lh * 8 + r);
                xg[row * GSZ + col] = acc[mt][nt][r] + bsum;
            }
        }
    }
}

// ---------------------------------------------------------------------------
// Kernel C: recurrent scan. 16 persistent workgroups; WG k owns hidden cols
// [k*32, k*32+32) -> gate cols {j, 512+j, 1024+j, 1536+j}. Its 512x128 bf16
// slice of W_hh lives in LDS (128KB) for the whole scan (possible thanks to
// the 320KB WGP LDS); per step, h (64x512 bf16) is async-copied into LDS in
// fragment order. Fragment LDS layout is [frag][chunk][lane][8 halfs]: each
// 16B chunk of 16 consecutive lanes stripes all 64 banks exactly once ->
// conflict-free ds_load_b128. c stays in registers in C/D-fragment layout so
// i/f/g/o of each (b,j) share a lane/VGPR slot. Cross-WG sync: device-scope
// atomic barrier (16 resident WGs); a cluster-barrier signal/wait pair is
// also emitted (HW NOP unless dispatched as a 16-WG cluster).
// ---------------------------------------------------------------------------
__global__ __launch_bounds__(256) void lstm_scan_kernel(
    const float* __restrict__ xg,             // [T*B][GSZ]
    const float* __restrict__ c0,             // [B][H]
    const unsigned short* __restrict__ whh,   // [GSZ][H] bf16
    unsigned short* __restrict__ hbuf,        // [2][B][H] bf16, buf0 = h0
    float* __restrict__ out,                  // [T][B][H]
    float* __restrict__ hT,                   // [B][H]
    float* __restrict__ cT,                   // [B][H]
    unsigned int* __restrict__ barrier_cnt)
{
    extern __shared__ char smem[];
    unsigned short* Wlds = (unsigned short*)smem;              // 128 frags * 256 halfs = 128KB
    unsigned short* hlds = (unsigned short*)(smem + 131072);   //  64 frags * 256 halfs =  64KB

    const int tid  = threadIdx.x;
    const int lane = tid & 31;
    const int wv   = tid >> 5;
    const int kwg  = blockIdx.x;      // hidden slice owner
    const int mt   = wv >> 1;         // batch tile: rows mt*16..mt*16+15
    const int jh   = wv & 1;          // hidden-column half (0 -> j 0..15, 1 -> 16..31)
    const int lh   = lane >> 4;
    const int ll   = lane & 15;
    const unsigned int nblocks = gridDim.x;

    // ---- preload W_hh slice into LDS, pre-swizzled into B-fragment order ----
    // Fragment f = nt*16+ks at halfs [f*256, f*256+256): chunk hi at +hi*128,
    // lane l at +l*8. Chunk hi of lane l = W[wrow][ks*32 + (l>>4)*16 + hi*8 ..+8).
    for (int cid = tid; cid < 8192; cid += 256) {
        const int hi = cid & 1;
        const int l  = (cid >> 1) & 31;
        const int ks = (cid >> 6) & 15;
        const int nt = cid >> 10;
        const int wrow = (nt >> 1) * HSZ + kwg * 32 + (nt & 1) * 16 + (l & 15);
        const size_t gofs = (size_t)wrow * HSZ + ks * 32 + (l >> 4) * 16 + hi * 8;
        const int dofs = (nt * 16 + ks) * 256 + hi * 128 + l * 8;
        copy16_g2l(whh + gofs, Wlds + dofs);
    }

    // ---- init cell state in registers (matches C/D fragment layout) ----
    const int jcol = kwg * 32 + jh * 16 + ll;   // this lane's hidden column
    float creg[8];
#pragma unroll
    for (int r = 0; r < 8; ++r) {
        const int b = mt * 16 + lh * 8 + r;
        creg[r] = c0[(size_t)b * HSZ + jcol];
    }

    unsigned int barrier_phase = 0;

    for (int t = 0; t < T_STEPS; ++t) {
        const unsigned short* hsrc = hbuf + (size_t)(t & 1) * (BATCH * HSZ);
        unsigned short* hdst = hbuf + (size_t)((t + 1) & 1) * (BATCH * HSZ);

        // prefetch next step's x_gates slab (this WG's 4 gate segments) so the
        // HBM read overlaps this step's WMMA work
        if (t + 1 < T_STEPS) {
            const int pb = tid & 63;       // batch row
            const int pg = tid >> 6;       // gate 0..3
            __builtin_prefetch(xg + ((size_t)(t + 1) * BATCH + pb) * GSZ +
                                   (size_t)pg * HSZ + kwg * 32, 0, 3);
        }

        // stage h into LDS in A-fragment order (async direct-to-LDS)
        for (int cid = tid; cid < 4096; cid += 256) {
            const int hi = cid & 1;
            const int l  = (cid >> 1) & 31;
            const int ks = (cid >> 6) & 15;
            const int m  = cid >> 10;
            const int row = m * 16 + (l & 15);
            const size_t gofs = (size_t)row * HSZ + ks * 32 + (l >> 4) * 16 + hi * 8;
            const int dofs = (m * 16 + ks) * 256 + hi * 128 + l * 8;
            copy16_g2l(hsrc + gofs, hlds + dofs);
        }
        g2l_wait();          // ASYNCcnt -> 0 : this thread's LDS writes landed
        __syncthreads();     // make them visible to all waves

        // ---- gates[:, kwg slice] = h @ Whh^T slice ----
        v8f acc[4] = {};
        for (int ks = 0; ks < 16; ++ks) {
            Frag16 a;
            const int abase = (mt * 16 + ks) * 256;
            a.u[0] = *(const uint4*)(hlds + abase + lane * 8);
            a.u[1] = *(const uint4*)(hlds + abase + 128 + lane * 8);
#pragma unroll
            for (int g = 0; g < 4; ++g) {
                const int nt = g * 2 + jh;
                Frag16 bf;
                const int bbase = (nt * 16 + ks) * 256;
                bf.u[0] = *(const uint4*)(Wlds + bbase + lane * 8);
                bf.u[1] = *(const uint4*)(Wlds + bbase + 128 + lane * 8);
                acc[g] = __builtin_amdgcn_wmma_f32_16x16x32_bf16(
                    false, a.v, false, bf.v, (short)0, acc[g], false, false);
            }
        }

        // ---- gate nonlinearity + state update + stores ----
#pragma unroll
        for (int r = 0; r < 8; ++r) {
            const int b = mt * 16 + lh * 8 + r;
            const size_t xbase = ((size_t)t * BATCH + b) * GSZ;
            const float gi = fsigmoid(acc[0][r] + xg[xbase + jcol]);
            const float gf = fsigmoid(acc[1][r] + xg[xbase + HSZ + jcol]);
            const float gg = ftanh  (acc[2][r] + xg[xbase + 2 * HSZ + jcol]);
            const float go = fsigmoid(acc[3][r] + xg[xbase + 3 * HSZ + jcol]);
            const float c  = gf * creg[r] + gi * gg;
            const float h  = go * ftanh(c);
            creg[r] = c;
            const size_t hofs = (size_t)b * HSZ + jcol;
            out[(size_t)t * (BATCH * HSZ) + hofs] = h;
            hdst[hofs] = f32_to_bf16(h);
            if (t == T_STEPS - 1) { hT[hofs] = h; cT[hofs] = c; }
        }

        // ---- grid-wide barrier (16 resident workgroups) ----
        __syncthreads();
#if HAVE_CLUSTER_BARRIER
        // HW NOP when ClusterID==0 (normal dispatch); real barrier if the
        // dispatch is ever configured as a 16-workgroup cluster.
        __builtin_amdgcn_s_cluster_barrier();
#endif
        if (tid == 0) {
            __threadfence();
            atomicAdd(barrier_cnt, 1u);
            barrier_phase += nblocks;
            while (__hip_atomic_load(barrier_cnt, __ATOMIC_ACQUIRE,
                                     __HIP_MEMORY_SCOPE_AGENT) < barrier_phase) { }
        }
        __syncthreads();
    }
}

// ---------------------------------------------------------------------------
extern "C" void kernel_launch(void* const* d_in, const int* in_sizes, int n_in,
                              void* d_out, int out_size, void* d_ws, size_t ws_size,
                              hipStream_t stream) {
    const float* input_    = (const float*)d_in[0];
    const float* h0        = (const float*)d_in[1];
    const float* c0        = (const float*)d_in[2];
    const float* weight_ih = (const float*)d_in[3];
    const float* weight_hh = (const float*)d_in[4];
    const float* bias_ih   = (const float*)d_in[5];
    const float* bias_hh   = (const float*)d_in[6];

    // workspace carve-up
    char* ws = (char*)d_ws;
    float* xg = (float*)ws;                                         // 512 MB
    size_t ofs = (size_t)T_STEPS * BATCH * GSZ * sizeof(float);
    unsigned short* xb   = (unsigned short*)(ws + ofs); ofs += (size_t)T_STEPS * BATCH * ISZ * 2;
    unsigned short* wih  = (unsigned short*)(ws + ofs); ofs += (size_t)GSZ * ISZ * 2;
    unsigned short* whh  = (unsigned short*)(ws + ofs); ofs += (size_t)GSZ * HSZ * 2;
    unsigned short* hbuf = (unsigned short*)(ws + ofs); ofs += (size_t)2 * BATCH * HSZ * 2;
    unsigned int*   cnt  = (unsigned int*)(ws + ofs);

    float* out = (float*)d_out;
    float* hT  = out + (size_t)T_STEPS * BATCH * HSZ;
    float* cT  = hT + (size_t)BATCH * HSZ;

    // Phase 0: bf16 conversions
    convert_f32_bf16_kernel<<<2048, 256, 0, stream>>>(input_, xb, T_STEPS * BATCH * ISZ);
    convert_f32_bf16_kernel<<<512, 256, 0, stream>>>(weight_ih, wih, GSZ * ISZ);
    convert_f32_bf16_kernel<<<512, 256, 0, stream>>>(weight_hh, whh, GSZ * HSZ);
    convert_f32_bf16_kernel<<<64, 256, 0, stream>>>(h0, hbuf, BATCH * HSZ);

    // Phase 1: x_gates GEMM (WMMA bf16)
    dim3 gB(512, 16);
    xgates_gemm_kernel<<<gB, 256, 0, stream>>>(xb, wih, bias_ih, bias_hh, xg);

    // Phase 2: recurrent scan (16 persistent WGs, 192KB dynamic LDS each)
    (void)hipMemsetAsync(cnt, 0, sizeof(unsigned int), stream);
    (void)hipFuncSetAttribute(reinterpret_cast<const void*>(&lstm_scan_kernel),
                              hipFuncAttributeMaxDynamicSharedMemorySize, 196608);
    lstm_scan_kernel<<<16, 256, 196608, stream>>>(xg, c0, whh, hbuf, out, hT, cT, cnt);
}
